// Block_29214367548032
// MI455X (gfx1250) — matrix operand order repaired and checked by reading, not compiled, not run on hardware
//
#include <hip/hip_runtime.h>

// ---------------------------------------------------------------------------
// Types
// ---------------------------------------------------------------------------
typedef __bf16 bf16;
typedef __bf16 bf16x16 __attribute__((ext_vector_type(16)));
typedef float  f32x8   __attribute__((ext_vector_type(8)));
typedef unsigned int u32x4 __attribute__((ext_vector_type(4)));
typedef int v4i __attribute__((__vector_size__(16)));   // int4, matches builtin param type

#define LDH 72  // LDS row pitch in halfs (64 data + 8 pad) -> 144B, 16B aligned, conflict-free

// ---------------------------------------------------------------------------
// CDNA5 async global->LDS copy path (ASYNCcnt), with plain-copy fallback.
// Builtin signature (from hipcc diagnostic): (v4i AS1*, v4i AS3*, imm, imm)
// ---------------------------------------------------------------------------
#if __has_builtin(__builtin_amdgcn_global_load_async_to_lds_b128)
#define GEMM_ASYNC 1
#if __has_builtin(__builtin_amdgcn_s_wait_asynccnt)
#define ASYNC_WAIT0() __builtin_amdgcn_s_wait_asynccnt(0)
#else
#define ASYNC_WAIT0() asm volatile("s_wait_asynccnt 0x0" ::: "memory")
#endif
__device__ __forceinline__ void copy16_to_lds(void* lp, const void* gp) {
    __builtin_amdgcn_global_load_async_to_lds_b128(
        (__attribute__((address_space(1))) v4i*)gp,
        (__attribute__((address_space(3))) v4i*)lp, 0, 0);
}
#else
#define GEMM_ASYNC 0
#define ASYNC_WAIT0()
__device__ __forceinline__ void copy16_to_lds(void* lp, const void* gp) {
    *(u32x4*)lp = *(const u32x4*)gp;
}
#endif

// ---------------------------------------------------------------------------
// WMMA fragment loaders (CDNA5 16x16x32 bf16 layouts, ISA 7.12.2)
// A 16x32: lanes 0-15 row=lane, elems = K{0..7},{16..23}; lanes 16-31 K{8..15},{24..31}
// B 32x16: lane = column, lanes 0-15 elems = K{0..7},{16..23}; lanes 16-31 K{8..15},{24..31}
//          (B stored transposed in LDS: row = output column n, contiguous K)
// ---------------------------------------------------------------------------
__device__ __forceinline__ bf16x16 frag_a(const bf16* t, int ks, int lane) {
    const bf16* p = t + (lane & 15) * LDH + ks * 32 + ((lane & 16) ? 8 : 0);
    union { bf16x16 v; u32x4 q[2]; } u;
    u.q[0] = *(const u32x4*)(p);
    u.q[1] = *(const u32x4*)(p + 16);
    return u.v;
}
__device__ __forceinline__ bf16x16 frag_b(const bf16* t, int ks, int lane) {
    const bf16* p = t + (lane & 15) * LDH + ks * 32 + ((lane & 16) ? 8 : 0);
    union { bf16x16 v; u32x4 q[2]; } u;
    u.q[0] = *(const u32x4*)(p);
    u.q[1] = *(const u32x4*)(p + 16);
    return u.v;
}
__device__ __forceinline__ f32x8 wmma_bf16(bf16x16 a, bf16x16 b, f32x8 c) {
    return __builtin_amdgcn_wmma_f32_16x16x32_bf16(false, a, false, b, (short)0, c, false, false);
}

// ---------------------------------------------------------------------------
// Weight convert f32 -> bf16 (weights are already [N][K] = B^T layout)
// ---------------------------------------------------------------------------
__global__ void f32_to_bf16_kernel(const float* __restrict__ s, bf16* __restrict__ d, int n) {
    int i = blockIdx.x * 256 + threadIdx.x;
    if (i < n) d[i] = (bf16)s[i];
}

// ---------------------------------------------------------------------------
// Precompute rel-pos bias [6][49][49] and shift mask [16][49][49]
// ---------------------------------------------------------------------------
__device__ __forceinline__ int region28(int a) { return a < 21 ? 0 : (a < 25 ? 1 : 2); }

__global__ void build_bias_mask_kernel(const float* __restrict__ table,
                                       float* __restrict__ bias6,
                                       float* __restrict__ mask16) {
    int i = blockIdx.x * 256 + threadIdx.x;
    if (i < 6 * 2401) {
        int h = i / 2401, rem = i % 2401, n = rem / 49, m = rem % 49;
        int idx = (n / 7 - m / 7 + 6) * 13 + (n % 7 - m % 7 + 6);
        bias6[i] = table[idx * 6 + h];
    }
    if (i < 16 * 2401) {
        int wdx = i / 2401, rem = i % 2401, n = rem / 49, m = rem % 49;
        int hi = wdx >> 2, wi = wdx & 3;
        int cn = 3 * region28(hi * 7 + n / 7) + region28(wi * 7 + n % 7);
        int cm = 3 * region28(hi * 7 + m / 7) + region28(wi * 7 + m % 7);
        mask16[i] = (cn == cm) ? 0.0f : -100.0f;
    }
}

// ---------------------------------------------------------------------------
// LN1 (channel-dim, unbiased var) + roll(-3,-3) + window partition -> bf16 [tok][384]
// ---------------------------------------------------------------------------
__global__ __launch_bounds__(256) void ln1_kernel(const float* __restrict__ x,
                                                  const float* __restrict__ w,
                                                  const float* __restrict__ bi,
                                                  bf16* __restrict__ xn) {
    int i = blockIdx.x * 256 + threadIdx.x;   // 50176 spatial tokens
    int b = i / 784, p = i % 784;
    const float* xb = x + (size_t)b * 384 * 784 + p;
    float s = 0.f, ss = 0.f;
    for (int c = 0; c < 384; ++c) { float v = xb[(size_t)c * 784]; s += v; ss += v * v; }
    float mean = s * (1.0f / 384.0f);
    float var = (ss - s * mean) * (1.0f / 383.0f);
    float rstd = rsqrtf(var + 1e-5f);
    int h = p / 28, wq = p % 28;
    int h2 = h + 25; if (h2 >= 28) h2 -= 28;   // (h-3) mod 28
    int w2 = wq + 25; if (w2 >= 28) w2 -= 28;
    int tok = ((b * 4 + h2 / 7) * 4 + (w2 / 7)) * 49 + (h2 % 7) * 7 + (w2 % 7);
    bf16* orow = xn + (size_t)tok * 384;
    for (int c = 0; c < 384; ++c)
        orow[c] = (bf16)((xb[(size_t)c * 784] - mean) * rstd * w[c] + bi[c]);
}

// ---------------------------------------------------------------------------
// LN2 over token rows of fp32 [p][384] -> bf16 [p][384] (wave-per-token)
// ---------------------------------------------------------------------------
__global__ __launch_bounds__(256) void ln2_kernel(const float* __restrict__ x2,
                                                  const float* __restrict__ w,
                                                  const float* __restrict__ bi,
                                                  bf16* __restrict__ xn2) {
    int tok = blockIdx.x * 8 + (threadIdx.x >> 5);
    int lane = threadIdx.x & 31;
    const float* row = x2 + (size_t)tok * 384;
    float r[12]; float s = 0.f, ss = 0.f;
#pragma unroll
    for (int i = 0; i < 12; ++i) { float v = row[lane + 32 * i]; r[i] = v; s += v; ss += v * v; }
#pragma unroll
    for (int o = 16; o; o >>= 1) { s += __shfl_xor(s, o, 32); ss += __shfl_xor(ss, o, 32); }
    float mean = s * (1.0f / 384.0f);
    float var = (ss - s * mean) * (1.0f / 383.0f);
    float rstd = rsqrtf(var + 1e-5f);
    bf16* orow = xn2 + (size_t)tok * 384;
#pragma unroll
    for (int i = 0; i < 12; ++i) {
        int c = lane + 32 * i;
        orow[c] = (bf16)((r[i] - mean) * rstd * w[c] + bi[c]);
    }
}

// ---------------------------------------------------------------------------
// Generic bf16 WMMA GEMM: C[M,N] = A[M,K] * Bw^T  (Bw stored [N][K])
// 128x128 block tile, BK=64, 256 threads = 8 waves, wave = 16-row stripe x 8 tiles.
// Double-buffered LDS tiles filled with GLOBAL_LOAD_ASYNC_TO_LDS_B128: the next
// tile streams in (ASYNCcnt) while WMMAs consume the current one.
// MODE 0: bf16 store (QKV)        MODE 1: window-reverse+unshift + residual -> f32 x2
// MODE 2: exact GELU -> bf16      MODE 3: residual + transpose to BCHW f32 (d_out)
// ---------------------------------------------------------------------------
template <int MODE, int KD, int NCOL>
__global__ __launch_bounds__(256) void gemm_bf16_kernel(const bf16* __restrict__ A,
                                                        const bf16* __restrict__ Bw,
                                                        void* __restrict__ outp,
                                                        const float* __restrict__ aux) {
    constexpr int BM = 128, BK = 64;
    __shared__ bf16 As[2][BM][LDH];
    __shared__ bf16 Bs[2][BM][LDH];
    const int tid = threadIdx.x;
    const int lane = tid & 31;
    const int wave = tid >> 5;
    const int m0 = blockIdx.y * BM;
    const int n0 = blockIdx.x * BM;

    f32x8 acc[8];
#pragma unroll
    for (int t = 0; t < 8; ++t)
#pragma unroll
        for (int j = 0; j < 8; ++j) acc[t][j] = 0.f;

    auto issue_tile = [&](int k0, int buf) {
#pragma unroll
        for (int i = 0; i < 4; ++i) {
            int idx = tid + i * 256;           // 1024 16B-chunks per tile
            int r = idx >> 3, cg = idx & 7;
            copy16_to_lds(&As[buf][r][cg * 8], A + (size_t)(m0 + r) * KD + k0 + cg * 8);
            copy16_to_lds(&Bs[buf][r][cg * 8], Bw + (size_t)(n0 + r) * KD + k0 + cg * 8);
        }
    };

    issue_tile(0, 0);
    int cur = 0;
    for (int k0 = 0; k0 < KD; k0 += BK) {
        ASYNC_WAIT0();          // this wave's async fills of buf `cur` are done
        __syncthreads();        // ... and everyone else's too
        if (k0 + BK < KD) issue_tile(k0 + BK, cur ^ 1);   // stream next tile
#pragma unroll
        for (int ks = 0; ks < 2; ++ks) {
            bf16x16 af = frag_a(&As[cur][wave * 16][0], ks, lane);
#pragma unroll
            for (int t = 0; t < 8; ++t) {
                bf16x16 bfr = frag_b(&Bs[cur][t * 16][0], ks, lane);
                acc[t] = wmma_bf16(af, bfr, acc[t]);
            }
        }
        __syncthreads();        // everyone done reading buf `cur` before it is refilled
        cur ^= 1;
    }

    const int rl = (lane & 16) ? 8 : 0;
    const int cl = lane & 15;
#pragma unroll
    for (int t = 0; t < 8; ++t) {
#pragma unroll
        for (int j = 0; j < 8; ++j) {
            int row = m0 + wave * 16 + j + rl;
            int col = n0 + t * 16 + cl;
            float v = acc[t][j];
            if (MODE == 0) {
                ((bf16*)outp)[(size_t)row * NCOL + col] = (bf16)v;
            } else if (MODE == 1) {
                int win = row / 49, n = row - win * 49;
                int b = win >> 4, hi = (win >> 2) & 3, wi = win & 3;
                int r7 = n / 7, c7 = n - r7 * 7;
                int h = hi * 7 + r7 + 3; if (h >= 28) h -= 28;   // roll +3
                int w = wi * 7 + c7 + 3; if (w >= 28) w -= 28;
                size_t p = (size_t)b * 784 + h * 28 + w;
                float sc = aux[((size_t)b * 384 + col) * 784 + h * 28 + w];  // shortcut (BCHW)
                ((float*)outp)[p * 384 + col] = sc + v;
            } else if (MODE == 2) {
                float g = 0.5f * v * (1.0f + erff(v * 0.70710678118654752f));
                ((bf16*)outp)[(size_t)row * NCOL + col] = (bf16)g;
            } else {
                int b = row / 784, hw = row - b * 784;
                ((float*)outp)[((size_t)b * 384 + col) * 784 + hw] =
                    aux[(size_t)row * 384 + col] + v;
            }
        }
    }
}

// ---------------------------------------------------------------------------
// Attention: one block per (window, head), 128 threads (4 waves)
// S = (q.k) * 0.125 + bias + mask ; softmax ; O = P.V^T  -> bf16 [tok][h*64+d]
// bias+mask prefilled into Sf during the (async) load phase so the post-WMMA
// step is a branch-free unconditional accumulate.
// ---------------------------------------------------------------------------
__global__ __launch_bounds__(128) void attn_kernel(const bf16* __restrict__ qkv,
                                                   const float* __restrict__ bias6,
                                                   const float* __restrict__ mask16,
                                                   bf16* __restrict__ outp) {
    __shared__ bf16 Qs[64][LDH];
    __shared__ bf16 Ks[64][LDH];
    __shared__ bf16 Vt[64][LDH];   // transposed: row = d, col = m
    __shared__ float Sf[64][66];
    __shared__ bf16 Ps[64][LDH];

    const int tid = threadIdx.x, lane = tid & 31, wave = tid >> 5;
    const int win = blockIdx.x, head = blockIdx.y;
    const size_t base = (size_t)win * 49 * 1152 + (size_t)head * 64;
    const float* bh = bias6 + head * 2401;
    const float* mk = mask16 + (win & 15) * 2401;

    // Q, K rows via async-to-LDS (zero-pad rows 49..63 with direct stores)
    for (int idx = tid; idx < 64 * 8; idx += 128) {
        int r = idx >> 3, cg = idx & 7;
        if (r < 49) {
            copy16_to_lds(&Qs[r][cg * 8], qkv + base + (size_t)r * 1152 + cg * 8);
            copy16_to_lds(&Ks[r][cg * 8], qkv + base + 384 + (size_t)r * 1152 + cg * 8);
        } else {
            u32x4 z = {0u, 0u, 0u, 0u};
            *(u32x4*)&Qs[r][cg * 8] = z;
            *(u32x4*)&Ks[r][cg * 8] = z;
        }
    }
    // V transposed (zero-pad m 49..63) + prefill Sf with bias+mask / -1e30
    for (int idx = tid; idx < 64 * 64; idx += 128) {
        int m = idx >> 6, d = idx & 63;
        bf16 v = (bf16)0.0f;
        if (m < 49) v = qkv[base + 768 + (size_t)m * 1152 + d];
        Vt[d][m] = v;
        float pre = -1e30f;
        if (m < 49 && d < 49) pre = bh[m * 49 + d] + mk[m * 49 + d];  // (n=m, col=d)
        Sf[m][d] = pre;
    }
    ASYNC_WAIT0();
    __syncthreads();

    // S = Q K^T   (K-dim = d = 64 -> 2 ksteps)
    f32x8 acc[4];
#pragma unroll
    for (int t = 0; t < 4; ++t)
#pragma unroll
        for (int j = 0; j < 8; ++j) acc[t][j] = 0.f;
#pragma unroll
    for (int ks = 0; ks < 2; ++ks) {
        bf16x16 af = frag_a(&Qs[wave * 16][0], ks, lane);
#pragma unroll
        for (int t = 0; t < 4; ++t) {
            bf16x16 bfr = frag_b(&Ks[t * 16][0], ks, lane);
            acc[t] = wmma_bf16(af, bfr, acc[t]);
        }
    }
    const int rl = (lane & 16) ? 8 : 0;
    const int cl = lane & 15;
#pragma unroll
    for (int t = 0; t < 4; ++t)
#pragma unroll
        for (int j = 0; j < 8; ++j) {
            int n = wave * 16 + j + rl, m = t * 16 + cl;
            Sf[n][m] += acc[t][j] * 0.125f;   // pads stay -1e30 (acc == 0 there)
        }
    __syncthreads();

    // row softmax (threads 0..63, one row each), write bf16 probs (pad cols 49..63 = 0)
    if (tid < 64) {
        float mx = -1e30f;
        for (int m = 0; m < 49; ++m) mx = fmaxf(mx, Sf[tid][m]);
        float sum = 0.f;
        for (int m = 0; m < 49; ++m) sum += __expf(Sf[tid][m] - mx);
        float inv = (sum > 0.f) ? (1.0f / sum) : 0.f;
        for (int m = 0; m < 64; ++m)
            Ps[tid][m] = (m < 49) ? (bf16)(__expf(Sf[tid][m] - mx) * inv) : (bf16)0.0f;
    }
    __syncthreads();

    // O = P V^T   (K-dim = m = 64 -> 2 ksteps, output cols = d)
    f32x8 oacc[4];
#pragma unroll
    for (int t = 0; t < 4; ++t)
#pragma unroll
        for (int j = 0; j < 8; ++j) oacc[t][j] = 0.f;
#pragma unroll
    for (int ks = 0; ks < 2; ++ks) {
        bf16x16 af = frag_a(&Ps[wave * 16][0], ks, lane);
#pragma unroll
        for (int t = 0; t < 4; ++t) {
            bf16x16 bfr = frag_b(&Vt[t * 16][0], ks, lane);
            oacc[t] = wmma_bf16(af, bfr, oacc[t]);
        }
    }
#pragma unroll
    for (int t = 0; t < 4; ++t)
#pragma unroll
        for (int j = 0; j < 8; ++j) {
            int n = wave * 16 + j + rl, d = t * 16 + cl;
            if (n < 49)
                outp[((size_t)win * 49 + n) * 384 + head * 64 + d] = (bf16)oacc[t][j];
        }
}

// ---------------------------------------------------------------------------
// Launch
// ---------------------------------------------------------------------------
extern "C" void kernel_launch(void* const* d_in, const int* in_sizes, int n_in,
                              void* d_out, int out_size, void* d_ws, size_t ws_size,
                              hipStream_t stream) {
    const float* x      = (const float*)d_in[0];
    const float* n1w    = (const float*)d_in[1];
    const float* n1b    = (const float*)d_in[2];
    const float* qkv_w  = (const float*)d_in[3];
    const float* reltab = (const float*)d_in[4];
    const float* proj_w = (const float*)d_in[5];
    const float* n2w    = (const float*)d_in[6];
    const float* n2b    = (const float*)d_in[7];
    const float* mlp_w1 = (const float*)d_in[8];
    const float* mlp_w3 = (const float*)d_in[9];

    char* ws = (char*)d_ws;
    size_t off = 0;
    auto alloc = [&](size_t bytes) {
        size_t o = off;
        off += (bytes + 255) & ~(size_t)255;
        return o;
    };
    const size_t TOK = 50176;
    bf16* Wqkv   = (bf16*)(ws + alloc((size_t)1152 * 384 * 2));
    bf16* Wproj  = (bf16*)(ws + alloc((size_t)384 * 384 * 2));
    bf16* Wm1    = (bf16*)(ws + alloc((size_t)1536 * 384 * 2));
    bf16* Wm2    = (bf16*)(ws + alloc((size_t)384 * 1536 * 2));
    float* bias6 = (float*)(ws + alloc((size_t)6 * 2401 * 4));
    float* mask16= (float*)(ws + alloc((size_t)16 * 2401 * 4));
    bf16* Xn     = (bf16*)(ws + alloc(TOK * 384 * 2));    // LN1 out; reused as LN2 out
    float* X2    = (float*)(ws + alloc(TOK * 384 * 4));   // residual after attention (f32)
    bf16* QKV    = (bf16*)(ws + alloc(TOK * 1152 * 2));   // qkv; overlaid by MLP hidden
    bf16* AO     = (bf16*)(ws + alloc(TOK * 384 * 2));    // attn out; contiguous after QKV
    bf16* Hh     = QKV;  // 50176 x 1536 bf16 == QKV region + AO region exactly

    // weights -> bf16 (already [N][K] layout)
    f32_to_bf16_kernel<<<(1152 * 384 + 255) / 256, 256, 0, stream>>>(qkv_w, Wqkv, 1152 * 384);
    f32_to_bf16_kernel<<<(384 * 384 + 255) / 256, 256, 0, stream>>>(proj_w, Wproj, 384 * 384);
    f32_to_bf16_kernel<<<(1536 * 384 + 255) / 256, 256, 0, stream>>>(mlp_w1, Wm1, 1536 * 384);
    f32_to_bf16_kernel<<<(384 * 1536 + 255) / 256, 256, 0, stream>>>(mlp_w3, Wm2, 384 * 1536);
    build_bias_mask_kernel<<<(16 * 2401 + 255) / 256, 256, 0, stream>>>(reltab, bias6, mask16);

    // LN1 + shift + window partition
    ln1_kernel<<<196, 256, 0, stream>>>(x, n1w, n1b, Xn);

    // QKV GEMM: [50176,384] x [384,1152]
    gemm_bf16_kernel<0, 384, 1152><<<dim3(9, 392), 256, 0, stream>>>(Xn, Wqkv, QKV, nullptr);

    // windowed attention
    attn_kernel<<<dim3(1024, 6), 128, 0, stream>>>(QKV, bias6, mask16, AO);

    // proj GEMM + window reverse + unshift + residual -> X2 (token-major f32)
    gemm_bf16_kernel<1, 384, 384><<<dim3(3, 392), 256, 0, stream>>>(AO, Wproj, X2, x);

    // LN2 (row-wise)
    ln2_kernel<<<6272, 256, 0, stream>>>(X2, n2w, n2b, Xn);

    // MLP1 + exact GELU -> hidden bf16 (overlays dead QKV+AO)
    gemm_bf16_kernel<2, 384, 1536><<<dim3(12, 392), 256, 0, stream>>>(Xn, Wm1, Hh, nullptr);

    // MLP2 + residual + transpose to BCHW f32 output
    gemm_bf16_kernel<3, 1536, 384><<<dim3(3, 392), 256, 0, stream>>>(Hh, Wm2, d_out, X2);
}
